// rpn_60404420051301
// MI455X (gfx1250) — matrix operand order repaired and checked by reading, not compiled, not run on hardware
//
#include <hip/hip_runtime.h>

// ---------------------------------------------------------------------------
// Types for CDNA5 WMMA / TDM
// ---------------------------------------------------------------------------
typedef __attribute__((ext_vector_type(16))) __bf16 v16bf;
typedef __attribute__((ext_vector_type(8)))  float  v8f;
typedef __attribute__((ext_vector_type(4)))  unsigned int v4u;
typedef __attribute__((ext_vector_type(8)))  int    v8i;
typedef __attribute__((ext_vector_type(4)))  int    v4i;

union FragBF { v16bf v; uint4 u[2]; };

#if defined(__gfx1250__) && __has_builtin(__builtin_amdgcn_tensor_load_to_lds)
#define HAVE_TDM 1
#else
#define HAVE_TDM 0
#endif

// ---------------------------------------------------------------------------
// Problem constants
// ---------------------------------------------------------------------------
#define C_IN    1024
#define NPIX    4096          // 64*64
#define C_MID   512
#define KTOT    9216          // 1024*9
#define NANCH   36864         // 4096*9
#define PRE_NMS 2000
#define POST_NMS 300
#define NSORT   65536
#define GNUM    20
#define POOLE   (300*1024*49) // feat_pool elements

// d_out float offsets
#define OFF_POOL 0
#define OFF_LOSS 15052800
#define OFF_PK   15052801
#define OFF_NK   15052833
#define OFF_CT   15052929
#define OFF_RT   15053057

// workspace byte offsets
#define O_XPAD  ((size_t)0)                    // 1024*66*66 bf16  = 8,921,088
#define O_WR    ((size_t)8921344)              // 512*9216 bf16    = 9,437,184
#define O_XACT  ((size_t)18358528)             // 512*4096 f32     = 8,388,608
#define O_SC2   ((size_t)26747136)             // 36864*2 f32
#define O_BB4   ((size_t)27042048)             // 36864*4 f32
#define O_PROP  ((size_t)27631872)             // 36864*4 f32
#define O_FG    ((size_t)28221696)             // 36864 f32
#define O_SKEY  ((size_t)28369152)             // 65536 f32
#define O_SIDX  ((size_t)28631296)             // 65536 i32
#define O_ROIS  ((size_t)28893440)             // 300*4 f32
#define O_GTMAX ((size_t)28898304)             // 20 f32
#define O_RED   ((size_t)28898560)             // 4*36864 f32

// ---------------------------------------------------------------------------
// Helpers
// ---------------------------------------------------------------------------
__device__ inline unsigned short f2bf(float f) {
  unsigned int u = __float_as_uint(f);
  unsigned int r = u + 0x7FFFu + ((u >> 16) & 1u);
  return (unsigned short)(r >> 16);
}

__device__ inline float iou_box(float a0,float a1,float a2,float a3,
                                float b0,float b1,float b2,float b3) {
  float lx = fmaxf(a0,b0), ly = fmaxf(a1,b1);
  float rx = fminf(a2,b2), ry = fminf(a3,b3);
  float w = fmaxf(rx-lx,0.f), h = fmaxf(ry-ly,0.f);
  float inter = w*h;
  float aa = (a2-a0)*(a3-a1), ab = (b2-b0)*(b3-b1);
  return inter / (aa + ab - inter + 1e-9f);
}

__device__ inline void encode_box(float a0,float a1,float a2,float a3,
                                  float g0,float g1,float g2,float g3,float* t) {
  float aw = a2-a0+1e-9f, ah = a3-a1+1e-9f;
  float acx = a0+0.5f*aw, acy = a1+0.5f*ah;
  float gw = fmaxf(g2-g0,1e-6f), gh = fmaxf(g3-g1,1e-6f);
  float gcx = g0+0.5f*gw, gcy = g1+0.5f*gh;
  t[0] = (gcx-acx)/aw; t[1] = (gcy-acy)/ah;
  t[2] = logf(gw/aw);  t[3] = logf(gh/ah);
}

__device__ inline void anchor_box(int idx, float* b) {
  int n  = idx / 9, a = idx - n*9;
  int ri = a / 3, si = a - ri*3;
  float r = 0.5f * (float)(1 << ri);     // 0.5,1,2
  float s = (float)(8 << si);            // 8,16,32
  float wsa = 16.0f * s * sqrtf(1.0f / r);
  float hsa = 16.0f * s * sqrtf(r);
  int y = n >> 6, x = n & 63;
  float cx = (x + 0.5f) * 16.0f, cy = (y + 0.5f) * 16.0f;
  b[0] = cx - 0.5f*wsa; b[1] = cy - 0.5f*hsa;
  b[2] = cx + 0.5f*wsa; b[3] = cy + 0.5f*hsa;
}

// ---------------------------------------------------------------------------
// K0: zero-pad + f32->bf16 input:  Xpad[1024][66][66]
// ---------------------------------------------------------------------------
__global__ void pad_input_kernel(const float* __restrict__ feat,
                                 unsigned short* __restrict__ xpad) {
  int e = blockIdx.x * 256 + threadIdx.x;
  if (e >= C_IN * 66 * 66) return;
  int ic = e / 4356; int rem = e - ic * 4356;
  int yy = rem / 66, xx = rem - yy * 66;
  float v = 0.f;
  if (yy >= 1 && yy <= 64 && xx >= 1 && xx <= 64)
    v = feat[(size_t)ic * NPIX + (yy-1)*64 + (xx-1)];
  xpad[e] = f2bf(v);
}

// ---------------------------------------------------------------------------
// K1: weight permute + f32->bf16:  Wr[m][k], k = kyx*1024 + ic
// ---------------------------------------------------------------------------
__global__ void perm_weight_kernel(const float* __restrict__ w_rpn,
                                   unsigned short* __restrict__ wr) {
  int e = blockIdx.x * 256 + threadIdx.x;
  if (e >= C_MID * KTOT) return;
  int m = e / KTOT; int k = e - m * KTOT;
  int kyx = k >> 10; int ic = k & 1023;
  float v = w_rpn[(size_t)(m * 1024 + ic) * 9 + kyx];
  wr[e] = f2bf(v);
}

// ---------------------------------------------------------------------------
// K2: implicit-GEMM 3x3 conv with bf16 WMMA + fused BN + ReLU
//     C[512][4096] = Wr[512][9216] x B[9216][4096]
//     block: 256 thr (8 waves); tile 128x128; K-step 32; 8 WMMA / wave / step
//     A tile via Tensor Data Mover (async DMA, TENSORcnt), B via coalesced
//     gather + register transpose.
// ---------------------------------------------------------------------------
__global__ __launch_bounds__(256)
void conv_gemm_kernel(const unsigned short* __restrict__ Wr,
                      const unsigned short* __restrict__ Xpad,
                      const float* __restrict__ b_rpn,
                      const float* __restrict__ bn_gamma,
                      const float* __restrict__ bn_beta,
                      const float* __restrict__ bn_mean,
                      const float* __restrict__ bn_var,
                      float* __restrict__ xact) {
  __shared__ uint4 As4[128 * 4];  // 128 rows x 32 bf16 (64B rows)
  __shared__ uint4 Bs4[128 * 4];  // 128 cols x 32 bf16, [n][k]

  const int tid   = threadIdx.x;
  const int lane  = tid & 31;
  const int wave  = tid >> 5;
  const int l15   = lane & 15;
  const int lHi   = lane >> 4;
  const int wm    = wave & 3;   // M quadrant (32 rows)
  const int wn    = wave >> 2;  // N half (64 cols)
  const int m0    = blockIdx.y * 128;
  const int n0    = blockIdx.x * 128;

  const uint4* Wr4 = (const uint4*)Wr;
  (void)Wr4;

#if HAVE_TDM
  // LDS byte offset of the A staging buffer (for the D# descriptor)
  unsigned ldsA = (unsigned)(unsigned long long)
      ((__attribute__((address_space(3))) char*)(&As4[0]));
#endif

  v8f acc[2][4] = {};

  for (int step = 0; step < 288; ++step) {
    const int k0     = step * 32;
    const int kyx    = k0 >> 10;
    const int icBase = k0 & 1023;
    const int ky     = kyx / 3;
    const int kx     = kyx - ky * 3;

    // ---- stage A: 128x32 bf16 tile of Wr
#if HAVE_TDM
    if (wave == 0) {
      unsigned long long ga =
          (unsigned long long)(const void*)Wr + ((size_t)m0 * KTOT + k0) * 2ull;
      v4u g0;
      g0[0] = 1u;                                     // count=1, user mode
      g0[1] = ldsA;                                   // lds_addr (bytes)
      g0[2] = (unsigned)ga;                           // global_addr[31:0]
      g0[3] = ((unsigned)(ga >> 32) & 0x1FFFFFFu) | (2u << 30); // addr hi, type=2
      v8i g1;
      g1[0] = 0x00010000;                             // data_size=1 (2B), no pad
      g1[1] = (int)((KTOT & 0xFFFF) << 16);           // tensor_dim0 lo16
      g1[2] = (int)((KTOT >> 16) | ((C_MID & 0xFFFF) << 16)); // dim0 hi, dim1 lo
      g1[3] = (int)((C_MID >> 16) | (32u << 16));     // dim1 hi, tile_dim0=32
      g1[4] = 128;                                    // tile_dim1=128, tile_dim2=0
      g1[5] = KTOT;                                   // tensor_dim0_stride lo32
      g1[6] = 0;
      g1[7] = 0;
      v4i z4 = {0,0,0,0};
#if __clang_major__ >= 23
      v8i z8 = {0,0,0,0,0,0,0,0};
      __builtin_amdgcn_tensor_load_to_lds(g0, g1, z4, z4, z8, 0);
#else
      __builtin_amdgcn_tensor_load_to_lds(g0, g1, z4, z4, 0);
#endif
    }
#else
    {
      int row = tid >> 1;
      int kk  = (tid & 1) * 16;
      const uint4* src = Wr4 + (size_t)(m0 + row) * 1152 + ((k0 + kk) >> 3);
      uint4 a0 = src[0];
      uint4 a1 = src[1];
      As4[row * 4 + (kk >> 3)]     = a0;
      As4[row * 4 + (kk >> 3) + 1] = a1;
    }
#endif

    // ---- stage B: gather 32(k) x 128(n) into LDS [n][k]
    //      consecutive lanes -> consecutive n (x) => each strided u16 load is
    //      a fully coalesced 64B wave access.
    #pragma unroll
    for (int jj = 0; jj < 2; ++jj) {
      int c   = tid + jj * 256;     // 0..511 chunks of 8
      int nl  = c & 127;            // local n
      int kc  = (c >> 7) << 3;      // 0,8,16,24
      int ng  = n0 + nl;
      int y   = ng >> 6, x = ng & 63;
      const unsigned short* base =
          Xpad + (size_t)(icBase + kc) * 4356 + (size_t)(y + ky) * 66 + (x + kx);
      unsigned short h[8];
      #pragma unroll
      for (int l = 0; l < 8; ++l) h[l] = base[(size_t)l * 4356];
      if (step + 1 < 288) __builtin_prefetch(base + 32 * 4356, 0, 1);
      uint4 pk;
      pk.x = (unsigned)h[0] | ((unsigned)h[1] << 16);
      pk.y = (unsigned)h[2] | ((unsigned)h[3] << 16);
      pk.z = (unsigned)h[4] | ((unsigned)h[5] << 16);
      pk.w = (unsigned)h[6] | ((unsigned)h[7] << 16);
      Bs4[nl * 4 + (kc >> 3)] = pk;
    }

#if HAVE_TDM
    if (wave == 0) __builtin_amdgcn_s_wait_tensorcnt(0);
#endif
    __syncthreads();

    // ---- fragments (b128 LDS loads, layouts per CDNA5 ISA 7.12.2)
    FragBF afr[2], bfr[4];
    #pragma unroll
    for (int mf = 0; mf < 2; ++mf) {
      int row = wm * 32 + mf * 16 + l15;
      afr[mf].u[0] = As4[row * 4 + lHi];       // K 0..7  / 8..15
      afr[mf].u[1] = As4[row * 4 + 2 + lHi];   // K 16..23 / 24..31
    }
    #pragma unroll
    for (int nf = 0; nf < 4; ++nf) {
      int col = wn * 64 + nf * 16 + l15;
      bfr[nf].u[0] = Bs4[col * 4 + lHi];         // K 0..7  / 8..15
      bfr[nf].u[1] = Bs4[col * 4 + 2 + lHi];     // K 16..23 / 24..31
    }
    #pragma unroll
    for (int mf = 0; mf < 2; ++mf)
      #pragma unroll
      for (int nf = 0; nf < 4; ++nf)
        acc[mf][nf] = __builtin_amdgcn_wmma_f32_16x16x32_bf16(
            false, afr[mf].v, false, bfr[nf].v, (short)0, acc[mf][nf], false, false);
    __syncthreads();
  }

  // ---- epilogue: bias + BN + ReLU, f32 store
  #pragma unroll
  for (int mf = 0; mf < 2; ++mf) {
    int mBase = m0 + wm * 32 + mf * 16 + lHi * 8;
    #pragma unroll
    for (int nf = 0; nf < 4; ++nf) {
      int n = n0 + wn * 64 + nf * 16 + l15;
      #pragma unroll
      for (int r = 0; r < 8; ++r) {
        int m = mBase + r;
        float v  = acc[mf][nf][r];
        float sc = bn_gamma[m] * rsqrtf(bn_var[m] + 1e-5f);
        v = (v + b_rpn[m] - bn_mean[m]) * sc + bn_beta[m];
        xact[(size_t)m * NPIX + n] = fmaxf(v, 0.0f);
      }
    }
  }
}

// ---------------------------------------------------------------------------
// K3: heads (1x1 convs) + softmax fg + anchor decode
// ---------------------------------------------------------------------------
__global__ __launch_bounds__(128)
void heads_kernel(const float* __restrict__ xact,
                  const float* __restrict__ w_cls, const float* __restrict__ b_cls,
                  const float* __restrict__ w_box, const float* __restrict__ b_box,
                  const float* __restrict__ im_info,
                  float* __restrict__ sc2, float* __restrict__ bb4,
                  float* __restrict__ prop, float* __restrict__ fg) {
  int t = blockIdx.x * 128 + threadIdx.x;
  if (t >= NANCH) return;
  int n = t / 9, a = t - n * 9;
  const float* wc0 = w_cls + (size_t)(2*a) * 512;
  const float* wc1 = wc0 + 512;
  const float* wb  = w_box + (size_t)(4*a) * 512;
  float s0 = b_cls[2*a], s1 = b_cls[2*a+1];
  float d0 = b_box[4*a], d1 = b_box[4*a+1], d2 = b_box[4*a+2], d3 = b_box[4*a+3];
  for (int c = 0; c < 512; ++c) {
    float xv = xact[(size_t)c * NPIX + n];
    s0 += xv * wc0[c];       s1 += xv * wc1[c];
    d0 += xv * wb[c];        d1 += xv * wb[512 + c];
    d2 += xv * wb[1024 + c]; d3 += xv * wb[1536 + c];
  }
  sc2[2*t] = s0; sc2[2*t+1] = s1;
  bb4[4*t] = d0; bb4[4*t+1] = d1; bb4[4*t+2] = d2; bb4[4*t+3] = d3;
  float mx = fmaxf(s0, s1);
  float e0 = __expf(s0 - mx), e1 = __expf(s1 - mx);
  fg[t] = e1 / (e0 + e1);
  float ab[4]; anchor_box(t, ab);
  float aw = ab[2]-ab[0], ah = ab[3]-ab[1];
  float acx = ab[0] + 0.5f*aw, acy = ab[1] + 0.5f*ah;
  float dw = fminf(fmaxf(d2, -4.135f), 4.135f);
  float dh = fminf(fmaxf(d3, -4.135f), 4.135f);
  float cx = d0*aw + acx, cy = d1*ah + acy;
  float w  = aw * __expf(dw), h = ah * __expf(dh);
  float imh = im_info[0] - 1.0f, imw = im_info[1] - 1.0f;
  prop[4*t+0] = fminf(fmaxf(cx - 0.5f*w, 0.f), imw);
  prop[4*t+1] = fminf(fmaxf(cy - 0.5f*h, 0.f), imh);
  prop[4*t+2] = fminf(fmaxf(cx + 0.5f*w, 0.f), imw);
  prop[4*t+3] = fminf(fmaxf(cy + 0.5f*h, 0.f), imh);
}

// ---------------------------------------------------------------------------
// K4: single-block bitonic sort (descending by score, index tie-break)
// ---------------------------------------------------------------------------
__global__ __launch_bounds__(1024)
void sort_kernel(const float* __restrict__ fg, float* skey, int* sidx) {
  int tid = threadIdx.x;
  for (int e = tid; e < NSORT; e += 1024) {
    skey[e] = (e < NANCH) ? fg[e] : -1e30f;
    sidx[e] = e;
  }
  __threadfence(); __syncthreads();
  for (unsigned k = 2; k <= (unsigned)NSORT; k <<= 1) {
    for (unsigned j = k >> 1; j > 0; j >>= 1) {
      for (int e = tid; e < NSORT; e += 1024) {
        int ixj = e ^ (int)j;
        if (ixj > e) {
          float ka = skey[e], kb = skey[ixj];
          int   ia = sidx[e], ib = sidx[ixj];
          bool descBlock = ((e & (int)k) == 0);
          bool aBeforeB  = (ka > kb) || (ka == kb && ia < ib);
          bool doswap    = descBlock ? !aBeforeB : aBeforeB;
          if (doswap) { skey[e]=kb; skey[ixj]=ka; sidx[e]=ib; sidx[ixj]=ia; }
        }
      }
      __threadfence(); __syncthreads();
    }
  }
}

// ---------------------------------------------------------------------------
// K5: NMS over top-2000, output 300 ROIs
// ---------------------------------------------------------------------------
__global__ __launch_bounds__(1024)
void nms_kernel(const float* __restrict__ prop, const int* __restrict__ sidx,
                float* __restrict__ rois) {
  __shared__ float4 bx[PRE_NMS];
  __shared__ int keep[PRE_NMS];
  int tid = threadIdx.x;
  for (int i = tid; i < PRE_NMS; i += 1024) {
    int id = sidx[i];
    bx[i] = make_float4(prop[4*id], prop[4*id+1], prop[4*id+2], prop[4*id+3]);
    keep[i] = 1;
  }
  __syncthreads();
  for (int i = 0; i < PRE_NMS; ++i) {
    if (keep[i]) {
      float4 bi = bx[i];
      for (int j = i + 1 + tid; j < PRE_NMS; j += 1024) {
        if (keep[j]) {
          float4 bj = bx[j];
          if (iou_box(bi.x,bi.y,bi.z,bi.w, bj.x,bj.y,bj.z,bj.w) > 0.7f)
            keep[j] = 0;
        }
      }
    }
    __syncthreads();
  }
  if (tid == 0) {
    int cnt = 0;
    for (int i = 0; i < PRE_NMS && cnt < POST_NMS; ++i)
      if (keep[i]) { rois[4*cnt]=bx[i].x; rois[4*cnt+1]=bx[i].y;
                     rois[4*cnt+2]=bx[i].z; rois[4*cnt+3]=bx[i].w; ++cnt; }
    for (int i = 0; i < PRE_NMS && cnt < POST_NMS; ++i)
      if (!keep[i]) { rois[4*cnt]=bx[i].x; rois[4*cnt+1]=bx[i].y;
                      rois[4*cnt+2]=bx[i].z; rois[4*cnt+3]=bx[i].w; ++cnt; }
  }
}

// ---------------------------------------------------------------------------
// K6: ROI max-pool -> d_out[0..POOLE)
// ---------------------------------------------------------------------------
__global__ __launch_bounds__(256)
void roipool_kernel(const float* __restrict__ feat, const float* __restrict__ rois,
                    float* __restrict__ out) {
  int idx = blockIdx.x * 256 + threadIdx.x;
  if (idx >= POOLE) return;
  int r = idx / (1024 * 49);
  int rem = idx - r * 1024 * 49;
  int c = rem / 49;
  int p = rem - c * 49;
  int py = p / 7, px = p - py * 7;
  float x1 = rois[4*r+0]*0.0625f, y1 = rois[4*r+1]*0.0625f;
  float x2 = rois[4*r+2]*0.0625f, y2 = rois[4*r+3]*0.0625f;
  float best = -3.4e38f;
  #pragma unroll
  for (int sy = 0; sy < 2; ++sy)
    #pragma unroll
    for (int sx = 0; sx < 2; ++sx) {
      int jj = py*2 + sy, ii = px*2 + sx;
      float ys = y1 + ((jj + 0.5f) / 14.0f) * (y2 - y1);
      float xs = x1 + ((ii + 0.5f) / 14.0f) * (x2 - x1);
      int iy = min(max((int)floorf(ys), 0), 63);
      int ix = min(max((int)floorf(xs), 0), 63);
      best = fmaxf(best, feat[(size_t)c * NPIX + iy*64 + ix]);
    }
  out[idx] = best;
}

// ---------------------------------------------------------------------------
// K7a: per-gt max overlap over all anchors
// ---------------------------------------------------------------------------
__global__ __launch_bounds__(256)
void gtmax_kernel(const float* __restrict__ gtb, float* __restrict__ gtmax) {
  __shared__ float s[256];
  int g = blockIdx.x;
  const float* gb = gtb + (size_t)g * 5;
  float g0=gb[0], g1=gb[1], g2=gb[2], g3=gb[3];
  float best = -1.f;
  for (int i = threadIdx.x; i < NANCH; i += 256) {
    float ab[4]; anchor_box(i, ab);
    best = fmaxf(best, iou_box(ab[0],ab[1],ab[2],ab[3], g0,g1,g2,g3));
  }
  s[threadIdx.x] = best;
  __syncthreads();
  for (int off = 128; off > 0; off >>= 1) {
    if (threadIdx.x < off) s[threadIdx.x] = fmaxf(s[threadIdx.x], s[threadIdx.x + off]);
    __syncthreads();
  }
  if (threadIdx.x == 0) gtmax[g] = s[0];
}

// ---------------------------------------------------------------------------
// K7b: per-anchor loss terms (deterministic: written to arrays)
// ---------------------------------------------------------------------------
__global__ __launch_bounds__(256)
void rpn_target_kernel(const float* __restrict__ sc2, const float* __restrict__ bb4,
                       const float* __restrict__ gtb, const float* __restrict__ gtmax,
                       float* __restrict__ red) {
  int i = blockIdx.x * 256 + threadIdx.x;
  if (i >= NANCH) return;
  float ab[4]; anchor_box(i, ab);
  float best = -1.f; int arg = 0; int isbest = 0;
  for (int g = 0; g < GNUM; ++g) {
    const float* gb = gtb + (size_t)g * 5;
    float ov = iou_box(ab[0],ab[1],ab[2],ab[3], gb[0],gb[1],gb[2],gb[3]);
    if (ov > best) { best = ov; arg = g; }
    if (ov >= gtmax[g] - 1e-5f) isbest = 1;
  }
  int label = (best < 0.3f) ? 0 : -1;
  if (best >= 0.7f || isbest) label = 1;
  float nll = 0.f, validf = 0.f, sl1 = 0.f, posf = 0.f;
  if (label >= 0) {
    float v0 = sc2[2*i], v1 = sc2[2*i+1];
    float mx = fmaxf(v0, v1);
    float lse = mx + logf(expf(v0 - mx) + expf(v1 - mx));
    nll = lse - ((label == 1) ? v1 : v0);
    validf = 1.f;
  }
  if (label == 1) {
    const float* gb = gtb + (size_t)arg * 5;
    float tg[4]; encode_box(ab[0],ab[1],ab[2],ab[3], gb[0],gb[1],gb[2],gb[3], tg);
    #pragma unroll
    for (int c = 0; c < 4; ++c) {
      float dd = bb4[4*i + c] - tg[c];
      float ad = fabsf(dd);
      sl1 += (ad < 1.f) ? 0.5f*dd*dd : ad - 0.5f;
    }
    posf = 1.f;
  }
  red[i] = nll;  red[NANCH + i] = validf;
  red[2*NANCH + i] = sl1;  red[3*NANCH + i] = posf;
}

// ---------------------------------------------------------------------------
// K7c: fixed-order reduction -> rpn_loss
// ---------------------------------------------------------------------------
__global__ __launch_bounds__(1024)
void loss_reduce_kernel(const float* __restrict__ red, float* __restrict__ out) {
  __shared__ float s0[1024], s1[1024], s2[1024], s3[1024];
  int t = threadIdx.x;
  float a=0,b=0,c=0,d=0;
  for (int i = t; i < NANCH; i += 1024) {
    a += red[i]; b += red[NANCH+i]; c += red[2*NANCH+i]; d += red[3*NANCH+i];
  }
  s0[t]=a; s1[t]=b; s2[t]=c; s3[t]=d;
  __syncthreads();
  for (int off = 512; off > 0; off >>= 1) {
    if (t < off) { s0[t]+=s0[t+off]; s1[t]+=s1[t+off]; s2[t]+=s2[t+off]; s3[t]+=s3[t+off]; }
    __syncthreads();
  }
  if (t == 0) out[0] = s0[0]/fmaxf(s1[0],1.f) + s2[0]/fmaxf(s3[0],1.f);
}

// ---------------------------------------------------------------------------
// K8: proposal_target (pk, nk, ct, rt) -> d_out tail
// ---------------------------------------------------------------------------
__global__ __launch_bounds__(512)
void ptarget_kernel(const float* __restrict__ rois, const float* __restrict__ gtb,
                    float* __restrict__ out) {
  __shared__ float maxov[POST_NMS], fgsc[POST_NMS], bgsc[POST_NMS];
  __shared__ int argg[POST_NMS], used[POST_NMS];
  int t = threadIdx.x;
  if (t < POST_NMS) {
    float r0=rois[4*t], r1=rois[4*t+1], r2=rois[4*t+2], r3=rois[4*t+3];
    float best = -1.f; int bi = 0;
    for (int g = 0; g < GNUM; ++g) {
      const float* gb = gtb + (size_t)g * 5;
      float ov = iou_box(r0,r1,r2,r3, gb[0],gb[1],gb[2],gb[3]);
      if (ov > best) { best = ov; bi = g; }
    }
    maxov[t] = best; argg[t] = bi;
    fgsc[t] = (best >= 0.5f) ? best : -1.f;
    bgsc[t] = (best <  0.5f) ? best : -1.f;
    used[t] = 0;
  }
  __syncthreads();
  if (t == 0) {
    int pk[32], nk[96];
    for (int s = 0; s < 32; ++s) {
      float b = -1e30f; int bi = 0;
      for (int j = 0; j < POST_NMS; ++j)
        if (!used[j] && fgsc[j] > b) { b = fgsc[j]; bi = j; }
      pk[s] = bi; used[bi] = 1;
    }
    for (int j = 0; j < POST_NMS; ++j) used[j] = 0;
    for (int s = 0; s < 96; ++s) {
      float b = -1e30f; int bi = 0;
      for (int j = 0; j < POST_NMS; ++j)
        if (!used[j] && bgsc[j] > b) { b = bgsc[j]; bi = j; }
      nk[s] = bi; used[bi] = 1;
    }
    for (int s = 0; s < 32; ++s) out[OFF_PK + s] = (float)pk[s];
    for (int s = 0; s < 96; ++s) out[OFF_NK + s] = (float)nk[s];
    for (int s = 0; s < 32; ++s) {
      int rix = pk[s];
      bool fgv = fgsc[rix] > 0.f;
      out[OFF_CT + s] = fgv ? gtb[(size_t)argg[rix]*5 + 4] : 0.f;
      float tg[4] = {0.f,0.f,0.f,0.f};
      if (fgv) {
        const float* gb = gtb + (size_t)argg[rix] * 5;
        encode_box(rois[4*rix],rois[4*rix+1],rois[4*rix+2],rois[4*rix+3],
                   gb[0],gb[1],gb[2],gb[3], tg);
      }
      for (int c = 0; c < 4; ++c) out[OFF_RT + s*4 + c] = tg[c];
    }
    for (int s = 32; s < 128; ++s) out[OFF_CT + s] = 0.f;
  }
}

// ---------------------------------------------------------------------------
// launch
// ---------------------------------------------------------------------------
extern "C" void kernel_launch(void* const* d_in, const int* in_sizes, int n_in,
                              void* d_out, int out_size, void* d_ws, size_t ws_size,
                              hipStream_t stream) {
  const float* feat    = (const float*)d_in[0];
  const float* im_info = (const float*)d_in[1];
  const float* gtb     = (const float*)d_in[2];
  const float* w_rpn   = (const float*)d_in[3];
  const float* b_rpn   = (const float*)d_in[4];
  const float* gamma   = (const float*)d_in[5];
  const float* beta    = (const float*)d_in[6];
  const float* mean    = (const float*)d_in[7];
  const float* var     = (const float*)d_in[8];
  const float* w_cls   = (const float*)d_in[9];
  const float* b_cls   = (const float*)d_in[10];
  const float* w_box   = (const float*)d_in[11];
  const float* b_box   = (const float*)d_in[12];

  char* ws = (char*)d_ws;
  unsigned short* xpad = (unsigned short*)(ws + O_XPAD);
  unsigned short* wr   = (unsigned short*)(ws + O_WR);
  float* xact  = (float*)(ws + O_XACT);
  float* sc2   = (float*)(ws + O_SC2);
  float* bb4   = (float*)(ws + O_BB4);
  float* prop  = (float*)(ws + O_PROP);
  float* fg    = (float*)(ws + O_FG);
  float* skey  = (float*)(ws + O_SKEY);
  int*   sidx  = (int*)  (ws + O_SIDX);
  float* rois  = (float*)(ws + O_ROIS);
  float* gtmax = (float*)(ws + O_GTMAX);
  float* red   = (float*)(ws + O_RED);
  float* out   = (float*)d_out;

  pad_input_kernel<<<(C_IN*66*66 + 255)/256, 256, 0, stream>>>(feat, xpad);
  perm_weight_kernel<<<(C_MID*KTOT + 255)/256, 256, 0, stream>>>(w_rpn, wr);

  dim3 gemmGrid(NPIX/128, C_MID/128, 1);
  conv_gemm_kernel<<<gemmGrid, 256, 0, stream>>>(wr, xpad, b_rpn, gamma, beta,
                                                 mean, var, xact);

  heads_kernel<<<NANCH/128, 128, 0, stream>>>(xact, w_cls, b_cls, w_box, b_box,
                                              im_info, sc2, bb4, prop, fg);

  sort_kernel<<<1, 1024, 0, stream>>>(fg, skey, sidx);
  nms_kernel<<<1, 1024, 0, stream>>>(prop, sidx, rois);

  roipool_kernel<<<POOLE/256, 256, 0, stream>>>(feat, rois, out + OFF_POOL);

  gtmax_kernel<<<GNUM, 256, 0, stream>>>(gtb, gtmax);
  rpn_target_kernel<<<NANCH/256, 256, 0, stream>>>(sc2, bb4, gtb, gtmax, red);
  loss_reduce_kernel<<<1, 1024, 0, stream>>>(red, out + OFF_LOSS);

  ptarget_kernel<<<1, 512, 0, stream>>>(rois, gtb, out);
}
